// DeformConv3D_77790447665638
// MI455X (gfx1250) — matrix operand order repaired and checked by reading, not compiled, not run on hardware
//
#include <hip/hip_runtime.h>

#define LDIM 16
#define HDIM 96
#define WDIM 96
#define HW   (HDIM * WDIM)        // 9216
#define LHW  (LDIM * HDIM * WDIM) // 147456
#define CIN  64

#define ASTRIDE 72   // halo cin stride (padded from 64, keeps 16B alignment, spreads banks)
#define HALO_SP 612  // 3 (z) * 6 (y) * 34 (x) halo spatial points

typedef __attribute__((ext_vector_type(16))) __bf16 v16bf;
typedef __attribute__((ext_vector_type(8)))  float  v8f;

static __device__ __forceinline__ unsigned short f2bf(float f) {
  unsigned u = __float_as_uint(f);
  unsigned r = u + 0x7FFFu + ((u >> 16) & 1u);   // round-to-nearest-even
  return (unsigned short)(r >> 16);
}

union Frag { uint4 q[2]; v16bf v; };

static __device__ __forceinline__ v16bf ld_frag(const unsigned short* p0,
                                                const unsigned short* p1) {
  Frag f;
  f.q[0] = *(const uint4*)p0;
  f.q[1] = *(const uint4*)p1;
  return f.v;
}

// Transpose weights (Cout, Cin, 27) fp32 -> (27, Cout, Cin) bf16
__global__ __launch_bounds__(256)
void dprep_w(const float* __restrict__ w, unsigned short* __restrict__ wt, int Cout) {
  int idx = blockIdx.x * 256 + threadIdx.x;
  int total = Cout * CIN * 27;
  if (idx >= total) return;
  int co  = idx / (CIN * 27);
  int rem = idx - co * (CIN * 27);
  int ci  = rem / 27;
  int tap = rem - ci * 27;
  wt[((size_t)tap * Cout + co) * CIN + ci] = f2bf(w[idx]);
}

// Implicit-GEMM 3x3x3 conv, pad=1, stride=1.
// GEMM: M = Cout (A = weights, loaded as per-lane frags straight from L2-resident
// global), N = 128 spatial (B = input halo in LDS), K = 64*27.
// Block tile: 64 cout x (4y x 32x) spatial at fixed z. 8 waves, each 16 cout x 64 spatial.
// Halo staged once; no barriers inside the K loop.
__global__ __launch_bounds__(256)
void dconv_gemm_wmma(const float* __restrict__ A,            // (B, 64, L, H, W)
                     const unsigned short* __restrict__ Wt,  // (27, Cout, 64) bf16
                     const float* __restrict__ bias,         // (Cout) or null
                     float* __restrict__ out,                // (B, Cout, L, H, W)
                     int Cout) {
  __shared__ unsigned short halo[HALO_SP * ASTRIDE];  // 88128 B, staged once

  const int t  = threadIdx.x;
  const int wv = t >> 5;
  const int l  = t & 31;

  const int xbase   = blockIdx.x * 32;
  const int z       = blockIdx.y / 24;
  const int ybase   = (blockIdx.y % 24) * 4;
  const int nchunks = Cout >> 6;
  const int b       = blockIdx.z / nchunks;
  const int nblk    = blockIdx.z % nchunks;

  // ---- stage input halo once: bf16 [sp][cin] ----
  for (int i = t; i < HALO_SP * CIN; i += 256) {
    int cin = i / HALO_SP;
    int sp  = i - cin * HALO_SP;
    int rr  = sp / 34;
    int spx = sp - rr * 34;
    int spz = rr / 6;
    int spy = rr - spz * 6;
    int az = z + spz - 1;
    int ay = ybase + spy - 1;
    int ax = xbase + spx - 1;
    float v = 0.f;
    if ((unsigned)az < (unsigned)LDIM && (unsigned)ay < (unsigned)HDIM &&
        (unsigned)ax < (unsigned)WDIM)
      v = A[(((size_t)b * CIN + cin) * LDIM + az) * HW + ay * WDIM + ax];
    halo[sp * ASTRIDE + cin] = f2bf(v);
  }
  __syncthreads();

  v8f acc[4] = {};
  const int wy   = wv & 3;          // cout sub-tile (16 rows)
  const int wx   = wv >> 2;         // spatial half (64 cols)
  const int arow = 16 * wy + (l & 15);
  const int kba  = (l < 16) ? 0 : 8;   // A-frag K runs: [kba,kba+8) and [kba+16,kba+24)
  const int kbb  = (l < 16) ? 0 : 8;   // B-frag K runs: [kbb,kbb+8) and [kbb+16,kbb+24)
  const int ncol = l & 15;

  // Per-lane weight base: this lane's cout row, within this block's cout chunk.
  // A-frag for K-step kk comes straight from global (L2-resident, tiny).
  const unsigned short* wrow = Wt + ((size_t)(nblk * 64 + arow)) * CIN + kba;
  const size_t tapstride = (size_t)Cout * CIN;

#pragma unroll 2
  for (int kk = 0; kk < 54; ++kk) {   // 27 taps x 2 cin halves, K=32 each
    const int tap = kk >> 1;
    const int ch  = (kk & 1) * 32;
    const int dz  = tap / 9;
    const int rt  = tap - dz * 9;
    const int dy  = rt / 3;
    const int dx  = rt - dy * 3;

    const unsigned short* wp = wrow + (size_t)tap * tapstride + ch;
    v16bf afrag = ld_frag(wp, wp + 16);

#pragma unroll
    for (int j = 0; j < 4; ++j) {
      int n    = 64 * wx + 16 * j + ncol;
      int yrow = n >> 5;
      int xcol = n & 31;
      int sp   = (dz * 6 + yrow + dy) * 34 + (xcol + dx);
      const unsigned short* bb = &halo[sp * ASTRIDE + ch + kbb];
      v16bf bfrag = ld_frag(bb, bb + 16);
      acc[j] = __builtin_amdgcn_wmma_f32_16x16x32_bf16(
          false, afrag, false, bfrag, (short)0, acc[j], false, false);
    }
  }

  // ---- store: lanes 0-15 write 16 contiguous x floats -> coalesced ----
#pragma unroll
  for (int r = 0; r < 8; ++r) {
    int cg   = nblk * 64 + 16 * wy + r + 8 * (l >> 4);
    float bv = bias ? bias[cg] : 0.f;   // uniform branch, hoisted out of j loop
    float* orow = out + (((size_t)b * Cout + cg) * LDIM + z) * HW + ybase * WDIM + xbase;
#pragma unroll
    for (int j = 0; j < 4; ++j) {
      int n    = 64 * wx + 16 * j + ncol;
      int yrow = n >> 5;
      int xcol = n & 31;
      orow[yrow * WDIM + xcol] = acc[j][r] + bv;
    }
  }
}

// Trilinear deformable gather, matches reference th_batch_map_coordinates.
__global__ __launch_bounds__(256)
void dgather(const float* __restrict__ x, const float* __restrict__ off,
             float* __restrict__ def, int total) {
  int i = blockIdx.x * 256 + threadIdx.x;
  if (i >= total) return;
  int bc = i / LHW;
  int s  = i - bc * LHW;
  int zz = s / HW;
  int r2 = s - zz * HW;
  int yy = r2 / WDIM;
  int xx = r2 - yy * WDIM;
  int b  = bc >> 6;
  int c  = bc & 63;
  // off (B, 3C, L,H,W) viewed as (B*C, L,H,W, 3)
  const float* op = off + (size_t)b * (3 * CIN) * LHW + ((size_t)c * LHW + s) * 3;
  float cz = (float)zz + op[0];
  float cy = (float)yy + op[1];
  float cx = (float)xx + op[2];
  cz = fminf(fmaxf(cz, 0.f), (float)(LDIM - 1));
  cy = fminf(fmaxf(cy, 0.f), (float)(HDIM - 1));
  cx = fminf(fmaxf(cx, 0.f), (float)(WDIM - 1));
  float fz0 = floorf(cz), fy0 = floorf(cy), fx0 = floorf(cx);
  float fz1 = ceilf(cz),  fy1 = ceilf(cy),  fx1 = ceilf(cx);
  float dzf = cz - fz0, dyf = cy - fy0, dxf = cx - fx0;
  int z0 = (int)fz0, y0 = (int)fy0, x0 = (int)fx0;
  int z1 = (int)fz1, y1 = (int)fy1, x1 = (int)fx1;
  const float* xp = x + (size_t)bc * LHW;
#define G(zq, yq, xq) xp[((zq) * HDIM + (yq)) * WDIM + (xq)]
  float v = G(z0, y0, x0) * (1 - dxf) * (1 - dyf) * (1 - dzf)
          + G(z0, y0, x1) * dxf * (1 - dyf) * (1 - dzf)
          + G(z0, y1, x0) * (1 - dxf) * dyf * (1 - dzf)
          + G(z1, y0, x0) * (1 - dxf) * (1 - dyf) * dzf
          + G(z1, y0, x1) * dxf * (1 - dyf) * dzf
          + G(z1, y1, x0) * (1 - dxf) * dyf * dzf
          + G(z0, y1, x1) * dxf * dyf * (1 - dzf)
          + G(z1, y1, x1) * dxf * dyf * dzf;
#undef G
  def[i] = v;
}

extern "C" void kernel_launch(void* const* d_in, const int* in_sizes, int n_in,
                              void* d_out, int out_size, void* d_ws, size_t ws_size,
                              hipStream_t stream) {
  (void)in_sizes; (void)n_in; (void)out_size; (void)ws_size;
  const float* x      = (const float*)d_in[0];  // (2,64,16,96,96)
  const float* w_off  = (const float*)d_in[1];  // (192,64,3,3,3)
  const float* w_conv = (const float*)d_in[2];  // (64,64,3,3,3)
  const float* b_conv = (const float*)d_in[3];  // (64)
  float* out = (float*)d_out;                   // (2,64,16,96,96)

  // workspace layout
  unsigned short* w1t = (unsigned short*)d_ws;          // 27*192*64 bf16
  unsigned short* w2t = w1t + 27 * 192 * 64;            // 27*64*64 bf16
  float* off = (float*)(w2t + 27 * 64 * 64);            // 2*192*LHW fp32
  float* def = off + (size_t)2 * 192 * LHW;             // 2*64*LHW fp32

  dprep_w<<<(27 * 192 * 64 + 255) / 256, 256, 0, stream>>>(w_off, w1t, 192);
  dprep_w<<<(27 * 64 * 64 + 255) / 256, 256, 0, stream>>>(w_conv, w2t, 64);

  // conv1: x -> off (Cout=192, no bias). grid: 3 x-tiles, 16z*24y, 2 batch * 3 cout-chunks
  dconv_gemm_wmma<<<dim3(3, 384, 6), 256, 0, stream>>>(x, w1t, nullptr, off, 192);

  int total = 2 * CIN * LHW;
  dgather<<<(total + 255) / 256, 256, 0, stream>>>(x, off, def, total);

  // conv2: def -> out (Cout=64, +bias)
  dconv_gemm_wmma<<<dim3(3, 384, 2), 256, 0, stream>>>(def, w2t, b_conv, out, 64);
}
